// RelationLayer_36988258353784
// MI455X (gfx1250) — matrix operand order repaired and checked by reading, not compiled, not run on hardware
//
#include <hip/hip_runtime.h>

typedef __attribute__((ext_vector_type(16))) _Float16 v16h;
typedef __attribute__((ext_vector_type(8)))  _Float16 h8;
typedef __attribute__((ext_vector_type(8)))  float    v8f;
typedef __attribute__((ext_vector_type(4)))  float    f4;

#define DEVINL __device__ __forceinline__

DEVINL float lrelu(float x) { return x > 0.0f ? x : 0.1f * x; }

DEVINL v16h cat8(h8 lo, h8 hi) {
  return __builtin_shufflevector(lo, hi, 0,1,2,3,4,5,6,7,8,9,10,11,12,13,14,15);
}

// A fragment (16x32 f16). base -> row0 of the 16-row tile, row stride lda halfs.
// Per ISA: lanes 0-15 hold K = k0+[0..7] and k0+16+[0..7]; lanes 16-31 shift by 8.
DEVINL v16h load_a_frag(const _Float16* base, int lda, int lane, int k0) {
  int m  = lane & 15;
  int kh = (lane >> 4) << 3;                 // 0 or 8
  const _Float16* p = base + (size_t)m * lda + k0 + kh;
  h8 lo = *(const h8*)(p);
  h8 hi = *(const h8*)(p + 16);
  return cat8(lo, hi);
}

// B fragment (32x16 f16) for y = x @ W^T with row-major W (N x K): B[k][n] = W[n][k].
// Per lane: N = n0+(lane&15), K = k0+[0..7] and k0+16+[0..7]; lanes 16-31 shift by 8.
DEVINL v16h load_b_frag(const _Float16* W, int ldw, int lane, int n0, int k0) {
  int n  = n0 + (lane & 15);
  int ks = k0 + ((lane >> 4) << 3);          // 0 or 8
  const _Float16* p = W + (size_t)n * ldw + ks;
  h8 lo = *(const h8*)(p);
  h8 hi = *(const h8*)(p + 16);
  return cat8(lo, hi);
}

#define WMMA_F16(a, b, c) \
  __builtin_amdgcn_wmma_f32_16x16x32_f16(false, (a), false, (b), (short)0, (c), false, false)

// ---------------------------------------------------------------- prep kernels
__global__ void k_cvt_f16(const float* __restrict__ s, _Float16* __restrict__ d, int n) {
  int i = blockIdx.x * blockDim.x + threadIdx.x;
  if (i < n) d[i] = (_Float16)s[i];
}

// xp: (L+6, B, E) with 3-row zero halo top/bottom (max conv pad = 3)
__global__ void k_pad_x(const float* __restrict__ x, _Float16* __restrict__ xp) {
  int i = blockIdx.x * blockDim.x + threadIdx.x;
  const int n = 134 * 16 * 512;
  if (i >= n) return;
  int lp = i / (16 * 512);
  int rest = i - lp * (16 * 512);
  int l = lp - 3;
  xp[i] = (l < 0 || l >= 128) ? (_Float16)0.0f
                              : (_Float16)x[(size_t)l * (16 * 512) + rest];
}

// ----------------------------------------------------------------- conv branches
// f[l,b, br*64+c] = lrelu( sum_t sum_e xp[l+t-pad+3, b, e] * w[c, t*512+e] + cb[c] )
// GEMM: rows = 16 l's (fixed b), N = 64 channels (one 16-tile per wave), K = k*512.
__global__ void __launch_bounds__(128)
k_conv(const _Float16* __restrict__ xp, const _Float16* __restrict__ cwh,
       const float* __restrict__ cb0, const float* __restrict__ cb1,
       const float* __restrict__ cb2, const float* __restrict__ cb3,
       _Float16* __restrict__ fbuf) {
  int wg = blockIdx.x;                 // 512 = 4 branches * 16 b * 8 l-tiles
  int br = wg >> 7;
  int b  = (wg >> 3) & 15;
  int lt = wg & 7;
  int wave = threadIdx.x >> 5, lane = threadIdx.x & 31;
  int n0 = wave << 4;
  int k = (br << 1) + 1, pad = br;     // kernels {1,3,5,7}
  const _Float16* W = cwh + (size_t)32768 * br * br;  // offsets 0,1,4,9 * 32768
  int ldw = k << 9;
  int l0 = lt << 4;
  int m = lane & 15, kh = (lane >> 4) << 3;
  v8f acc = {};
  int nchunks = k << 4;                // (k*512)/32
  for (int c = 0; c < nchunks; ++c) {
    int t = c >> 4;
    int e0 = (c & 15) << 5;
    int row = l0 + m + t - pad + 3;
    const _Float16* pa = xp + ((size_t)row * 16 + b) * 512 + e0 + kh;
    h8 lo = *(const h8*)(pa);
    h8 hi = *(const h8*)(pa + 16);
    v16h a = cat8(lo, hi);
    v16h bf = load_b_frag(W, ldw, lane, n0, c << 5);
    acc = WMMA_F16(a, bf, acc);
  }
  const float* cb = br == 0 ? cb0 : br == 1 ? cb1 : br == 2 ? cb2 : cb3;
  int col = n0 + (lane & 15);
  float bias = cb[col];
  int rb = (lane >> 4) << 3;
#pragma unroll
  for (int r = 0; r < 8; ++r) {
    float vv = lrelu(acc[r] + bias);
    int l = l0 + rb + r;
    fbuf[((size_t)l * 16 + b) * 256 + (br << 6) + col] = (_Float16)vv;
  }
}

// -------------------------------------------------- u = f@Wa^T ; v = f@Wb^T + mb0
__global__ void __launch_bounds__(128)
k_uv(const _Float16* __restrict__ f, const _Float16* __restrict__ mw0h,
     const float* __restrict__ mb0, float* __restrict__ u, float* __restrict__ v) {
  int wave = threadIdx.x >> 5, lane = threadIdx.x & 31;
  int rt = blockIdx.x * 4 + wave;      // 0..127 row-tiles of (l*16+b)
  int r0 = rt << 4;
  const _Float16* A = f + (size_t)r0 * 256;
  v16h a[8];
#pragma unroll
  for (int c = 0; c < 8; ++c) a[c] = load_a_frag(A, 256, lane, c << 5);
  int col16 = lane & 15, rb = (lane >> 4) << 3;
#pragma unroll
  for (int half = 0; half < 2; ++half) {
    float* dst = half ? v : u;
    int koff = half << 8;              // Wa = mw0[:, :256], Wb = mw0[:, 256:]
#pragma unroll
    for (int nt = 0; nt < 4; ++nt) {
      v8f acc = {};
#pragma unroll
      for (int c = 0; c < 8; ++c) {
        v16h bf = load_b_frag(mw0h, 512, lane, nt << 4, koff + (c << 5));
        acc = WMMA_F16(a[c], bf, acc);
      }
      int col = (nt << 4) + col16;
      float bias = half ? mb0[col] : 0.0f;
#pragma unroll
      for (int r = 0; r < 8; ++r)
        dst[(size_t)(r0 + rb + r) * 64 + col] = acc[r] + bias;
    }
  }
}

// ------------------------------------------- pairwise MLP layers 1..3 + sum over i
// Two i-values per inner iteration so each LDS weight fragment feeds 2 WMMAs.
__global__ void __launch_bounds__(128)
k_pair(const float* __restrict__ u, const float* __restrict__ v,
       const _Float16* __restrict__ mwh,   // mw1,mw2,mw3 f16, contiguous 3*4096
       const float* __restrict__ mb1, const float* __restrict__ mb2,
       const float* __restrict__ mb3, _Float16* __restrict__ sh) {
  __shared__ __align__(16) _Float16 Wl[3 * 4096];      // 24 KB weights
  __shared__ __align__(16) _Float16 slab[8][16 * 64];  // 16 KB: 2 slabs per wave
  __shared__ __align__(16) float    red[4][16 * 64];   // 16 KB reduction
  int b = blockIdx.x, jt = blockIdx.y;
  int wave = threadIdx.x >> 5, lane = threadIdx.x & 31;
  {
    const uint4* src = (const uint4*)mwh;
    uint4* dst = (uint4*)Wl;
    for (int i = threadIdx.x; i < 3 * 4096 / 8; i += 128) dst[i] = src[i];
  }
  __syncthreads();
  int m = lane & 15, kh = (lane >> 4) << 3, col16 = lane & 15, rb = (lane >> 4) << 3;
  _Float16* slabA = slab[wave * 2];
  _Float16* slabB = slab[wave * 2 + 1];
  // u values held in registers, laid out to match A-fragment K groups
  float ur[32];
  {
    const float* up = u + (size_t)((jt * 16 + m) * 16 + b) * 64;
#pragma unroll
    for (int g = 0; g < 4; ++g) {
      f4 x0 = *(const f4*)(up + g * 16 + kh);
      f4 x1 = *(const f4*)(up + g * 16 + kh + 4);
#pragma unroll
      for (int e = 0; e < 4; ++e) { ur[g * 8 + e] = x0[e]; ur[g * 8 + 4 + e] = x1[e]; }
    }
  }
  float biasr[3][4];
#pragma unroll
  for (int nt = 0; nt < 4; ++nt) {
    int col = (nt << 4) + col16;
    biasr[0][nt] = mb1[col]; biasr[1][nt] = mb2[col]; biasr[2][nt] = mb3[col];
  }
  float asum[4][8];
#pragma unroll
  for (int nt = 0; nt < 4; ++nt)
#pragma unroll
    for (int r = 0; r < 8; ++r) asum[nt][r] = 0.0f;

  int i0 = wave * 32;                    // each wave covers 32 i values
  for (int ii = 0; ii < 32; ii += 2) {
    const float* vpA = v + (size_t)((i0 + ii) * 16 + b) * 64;
    const float* vpB = vpA + 16 * 64;
    if (ii + 2 < 32) {                   // prefetch next pair of v rows
      __builtin_prefetch(vpB + 16 * 64 + kh, 0, 0);
      __builtin_prefetch(vpB + 32 * 64 + kh, 0, 0);
    }
    v16h aA0, aA1, aB0, aB1;
#pragma unroll
    for (int g = 0; g < 4; ++g) {
      f4 xa0 = *(const f4*)(vpA + g * 16 + kh);
      f4 xa1 = *(const f4*)(vpA + g * 16 + kh + 4);
      f4 xb0 = *(const f4*)(vpB + g * 16 + kh);
      f4 xb1 = *(const f4*)(vpB + g * 16 + kh + 4);
      int e0 = (g & 1) << 3;             // element base within the fragment
#pragma unroll
      for (int e = 0; e < 4; ++e) {
        _Float16 ha0 = (_Float16)lrelu(ur[g * 8 + e] + xa0[e]);
        _Float16 ha1 = (_Float16)lrelu(ur[g * 8 + 4 + e] + xa1[e]);
        _Float16 hb0 = (_Float16)lrelu(ur[g * 8 + e] + xb0[e]);
        _Float16 hb1 = (_Float16)lrelu(ur[g * 8 + 4 + e] + xb1[e]);
        if (g < 2) { aA0[e0 + e] = ha0; aA0[e0 + 4 + e] = ha1;
                     aB0[e0 + e] = hb0; aB0[e0 + 4 + e] = hb1; }
        else       { aA1[e0 + e] = ha0; aA1[e0 + 4 + e] = ha1;
                     aB1[e0 + e] = hb0; aB1[e0 + 4 + e] = hb1; }
      }
    }
#pragma unroll
    for (int lay = 0; lay < 3; ++lay) {
#pragma unroll
      for (int nt = 0; nt < 4; ++nt) {
        v16h b0 = load_b_frag(&Wl[lay * 4096], 64, lane, nt << 4, 0);
        v16h b1 = load_b_frag(&Wl[lay * 4096], 64, lane, nt << 4, 32);
        v8f c0 = {}, c1 = {};
        c0 = WMMA_F16(aA0, b0, c0);
        c0 = WMMA_F16(aA1, b1, c0);
        c1 = WMMA_F16(aB0, b0, c1);
        c1 = WMMA_F16(aB1, b1, c1);
        if (lay < 2) {
#pragma unroll
          for (int r = 0; r < 8; ++r) {
            int idx = (rb + r) * 64 + (nt << 4) + col16;
            slabA[idx] = (_Float16)lrelu(c0[r] + biasr[lay][nt]);
            slabB[idx] = (_Float16)lrelu(c1[r] + biasr[lay][nt]);
          }
        } else {
#pragma unroll
          for (int r = 0; r < 8; ++r)
            asum[nt][r] += lrelu(c0[r] + biasr[2][nt]) +
                           lrelu(c1[r] + biasr[2][nt]);
        }
      }
      if (lay < 2) {                       // wave-private slabs: LDS is in-order
        aA0 = load_a_frag(slabA, 64, lane, 0);
        aA1 = load_a_frag(slabA, 64, lane, 32);
        aB0 = load_a_frag(slabB, 64, lane, 0);
        aB1 = load_a_frag(slabB, 64, lane, 32);
      }
    }
  }
#pragma unroll
  for (int nt = 0; nt < 4; ++nt)
#pragma unroll
    for (int r = 0; r < 8; ++r)
      red[wave][(rb + r) * 64 + (nt << 4) + col16] = asum[nt][r];
  __syncthreads();
  for (int idx = threadIdx.x; idx < 1024; idx += 128) {
    float sv = red[0][idx] + red[1][idx] + red[2][idx] + red[3][idx];
    int row = idx >> 6, col = idx & 63;
    sh[(size_t)((jt * 16 + row) * 16 + b) * 64 + col] = (_Float16)sv;
  }
}

// ------------------------------------------------- final 64->64 -> 64->256 layers
__global__ void __launch_bounds__(128)
k_final(const _Float16* __restrict__ sh, const _Float16* __restrict__ lw0h,
        const _Float16* __restrict__ lw1h, const float* __restrict__ lb0,
        const float* __restrict__ lb1, float* __restrict__ out) {
  __shared__ __align__(16) _Float16 W0[4096];
  __shared__ __align__(16) _Float16 W1[16384];
  __shared__ __align__(16) _Float16 slab[4][1024];
  int wave = threadIdx.x >> 5, lane = threadIdx.x & 31;
  {
    const uint4* s0 = (const uint4*)lw0h; uint4* d0 = (uint4*)W0;
    for (int i = threadIdx.x; i < 512; i += 128) d0[i] = s0[i];
    const uint4* s1 = (const uint4*)lw1h; uint4* d1 = (uint4*)W1;
    for (int i = threadIdx.x; i < 2048; i += 128) d1[i] = s1[i];
  }
  __syncthreads();
  int rt = blockIdx.x * 4 + wave;
  int r0 = rt << 4;
  int col16 = lane & 15, rb = (lane >> 4) << 3;
  v16h a0 = load_a_frag(sh + (size_t)r0 * 64, 64, lane, 0);
  v16h a1 = load_a_frag(sh + (size_t)r0 * 64, 64, lane, 32);
#pragma unroll
  for (int nt = 0; nt < 4; ++nt) {
    v8f c = {};
    v16h b0 = load_b_frag(W0, 64, lane, nt << 4, 0);
    v16h b1 = load_b_frag(W0, 64, lane, nt << 4, 32);
    c = WMMA_F16(a0, b0, c);
    c = WMMA_F16(a1, b1, c);
    int col = (nt << 4) + col16;
    float bias = lb0[col];
#pragma unroll
    for (int r = 0; r < 8; ++r)
      slab[wave][(rb + r) * 64 + col] = (_Float16)lrelu(c[r] + bias);
  }
  a0 = load_a_frag(slab[wave], 64, lane, 0);
  a1 = load_a_frag(slab[wave], 64, lane, 32);
  for (int nt = 0; nt < 16; ++nt) {
    v8f c = {};
    v16h b0 = load_b_frag(W1, 64, lane, nt << 4, 0);
    v16h b1 = load_b_frag(W1, 64, lane, nt << 4, 32);
    c = WMMA_F16(a0, b0, c);
    c = WMMA_F16(a1, b1, c);
    int col = (nt << 4) + col16;
    float bias = lb1[col];
#pragma unroll
    for (int r = 0; r < 8; ++r)
      out[(size_t)(r0 + rb + r) * 256 + col] = lrelu(c[r] + bias);
  }
}

// --------------------------------------------------------------------- launcher
extern "C" void kernel_launch(void* const* d_in, const int* in_sizes, int n_in,
                              void* d_out, int out_size, void* d_ws, size_t ws_size,
                              hipStream_t stream) {
  const float* x = (const float*)d_in[0];
  const float* cw[4] = {(const float*)d_in[1], (const float*)d_in[3],
                        (const float*)d_in[5], (const float*)d_in[7]};
  const float* cb[4] = {(const float*)d_in[2], (const float*)d_in[4],
                        (const float*)d_in[6], (const float*)d_in[8]};
  const float* mw0 = (const float*)d_in[9];
  const float* mb0 = (const float*)d_in[10];
  const float* mw1 = (const float*)d_in[11];
  const float* mb1 = (const float*)d_in[12];
  const float* mw2 = (const float*)d_in[13];
  const float* mb2 = (const float*)d_in[14];
  const float* mw3 = (const float*)d_in[15];
  const float* mb3 = (const float*)d_in[16];
  const float* lw0 = (const float*)d_in[17];
  const float* lb0 = (const float*)d_in[18];
  const float* lw1 = (const float*)d_in[19];
  const float* lb1 = (const float*)d_in[20];
  float* out = (float*)d_out;

  char* ws = (char*)d_ws;
  size_t o = 0;
  auto take = [&](size_t bytes) {
    size_t r = o;
    o = (o + bytes + 255) & ~(size_t)255;
    return r;
  };
  _Float16* xp   = (_Float16*)(ws + take((size_t)134 * 16 * 512 * 2)); // padded x f16
  _Float16* cwh  = (_Float16*)(ws + take((size_t)524288 * 2));         // conv wts f16
  _Float16* mw0h = (_Float16*)(ws + take((size_t)32768 * 2));
  _Float16* mwh  = (_Float16*)(ws + take((size_t)12288 * 2));          // mw1..3 f16
  _Float16* lw0h = (_Float16*)(ws + take((size_t)4096 * 2));
  _Float16* lw1h = (_Float16*)(ws + take((size_t)16384 * 2));
  _Float16* fbuf = (_Float16*)(ws + take((size_t)524288 * 2));         // f (L,B,256)
  _Float16* sh   = (_Float16*)(ws + take((size_t)131072 * 2));         // sum (L,B,64)
  float*    u    = (float*)   (ws + take((size_t)131072 * 4));
  float*    v    = (float*)   (ws + take((size_t)131072 * 4));
  // total ~5.8 MB of ws

  auto cvt = [&](const float* s, _Float16* d, int n) {
    k_cvt_f16<<<(n + 255) / 256, 256, 0, stream>>>(s, d, n);
  };
  const int cwoff[4] = {0, 32768, 131072, 294912};
  const int cwn[4]   = {32768, 98304, 163840, 229376};
  for (int i = 0; i < 4; ++i) cvt(cw[i], cwh + cwoff[i], cwn[i]);
  cvt(mw0, mw0h, 32768);
  cvt(mw1, mwh, 4096);
  cvt(mw2, mwh + 4096, 4096);
  cvt(mw3, mwh + 8192, 4096);
  cvt(lw0, lw0h, 4096);
  cvt(lw1, lw1h, 16384);
  {
    int n = 134 * 16 * 512;
    k_pad_x<<<(n + 255) / 256, 256, 0, stream>>>(x, xp);
  }
  k_conv<<<512, 128, 0, stream>>>(xp, cwh, cb[0], cb[1], cb[2], cb[3], fbuf);
  k_uv<<<32, 128, 0, stream>>>(fbuf, mw0h, mb0, u, v);
  dim3 g(16, 8);
  k_pair<<<g, 128, 0, stream>>>(u, v, mwh, mb1, mb2, mb3, sh);
  k_final<<<32, 128, 0, stream>>>(sh, lw0h, lw1h, lb0, lb1, out);
  (void)in_sizes; (void)n_in; (void)out_size; (void)ws_size;
}
